// cosine_seqNet_635655159821
// MI455X (gfx1250) — matrix operand order repaired and checked by reading, not compiled, not run on hardware
//
#include <hip/hip_runtime.h>
#include <hip/hip_bf16.h>

typedef __attribute__((ext_vector_type(16))) __bf16 v16bf;
typedef __attribute__((ext_vector_type(8)))  __bf16 v8bf;
typedef __attribute__((ext_vector_type(8)))  float  v8f;
typedef __attribute__((ext_vector_type(4)))  float  v4f;

constexpr int kB = 256;     // batch
constexpr int kS = 10;      // seq
constexpr int kD = 4096;    // in channels
constexpr int kO = 4096;    // out channels
constexpr int kW = 5;       // conv width
constexpr int kK = kD * kW; // 20480 GEMM reduction dim
constexpr float kEPS = 1e-8f;

constexpr int kNS = 10;     // LDS ring stages (32-K weight slices)
// ring row stride: 36 floats (144B) -> r*36 mod 64 distinct for all 16 rows
// => conflict-free ds_load_b128 consumption
constexpr int kRowF = 36;
constexpr int kStageBytes = 16 * kRowF * 4;   // 2304

// ---------------------------------------------------------------------------
// CDNA5 async / barrier helpers
// ---------------------------------------------------------------------------
__device__ __forceinline__ void async_load_b128(unsigned lds_byte_addr,
                                                const void* gaddr) {
    // GLOBAL_LOAD_ASYNC_TO_LDS_B128: LDS[vdst] = MEM[vaddr], tracked by ASYNCcnt
    asm volatile("global_load_async_to_lds_b128 %0, %1, off"
                 :: "v"(lds_byte_addr), "v"(gaddr)
                 : "memory");
}

__device__ __forceinline__ void wait_asynccnt_6() {
#if __has_builtin(__builtin_amdgcn_s_wait_asynccnt)
    __builtin_amdgcn_s_wait_asynccnt(6);
#else
    asm volatile("s_wait_asynccnt 0x6" ::: "memory");
#endif
}

// workgroup split-barrier WITHOUT the loadcnt/storecnt drain that
// __syncthreads() inserts (we only need LDS ordering; async writes are
// covered by s_wait_asynccnt, ds reads by the dscnt waits feeding the cvts)
__device__ __forceinline__ void block_barrier() {
    asm volatile("s_barrier_signal -1\n\t"
                 "s_barrier_wait -1" ::: "memory");
}

// ---------------------------------------------------------------------------
// Kernel 1: per-batch cosine weights.
// ---------------------------------------------------------------------------
__global__ __launch_bounds__(256) void stats_kernel(const float* __restrict__ x,
                                                    float* __restrict__ cosv) {
    const int b   = blockIdx.x;
    const int tid = threadIdx.x;
    const float* xb = x + (size_t)b * kS * kD;

    float dot[kS];
    float nrm[kS];
#pragma unroll
    for (int s = 0; s < kS; ++s) { dot[s] = 0.f; nrm[s] = 0.f; }

    for (int d = tid; d < kD; d += 256) {
        const float c = xb[6 * kD + d];
#pragma unroll
        for (int s = 0; s < kS; ++s) {
            const float v = xb[s * kD + d];
            dot[s] += v * c;
            nrm[s] += v * v;
        }
    }

#pragma unroll
    for (int s = 0; s < kS; ++s) {
#pragma unroll
        for (int off = 16; off > 0; off >>= 1) {
            dot[s] += __shfl_xor(dot[s], off, 32);
            nrm[s] += __shfl_xor(nrm[s], off, 32);
        }
    }

    __shared__ float sdot[kS][8];
    __shared__ float snrm[kS][8];
    const int wave = tid >> 5;
    const int lane = tid & 31;
    if (lane == 0) {
#pragma unroll
        for (int s = 0; s < kS; ++s) { sdot[s][wave] = dot[s]; snrm[s][wave] = nrm[s]; }
    }
    __syncthreads();

    if (tid < kS) {
        float ds = 0.f, ns = 0.f;
#pragma unroll
        for (int w = 0; w < 8; ++w) { ds += sdot[tid][w]; ns += snrm[tid][w]; }
        sdot[tid][0] = ds;
        snrm[tid][0] = ns;
    }
    __syncthreads();

    if (tid < kS) {
        const float cn = fmaxf(sqrtf(snrm[6][0]), kEPS);
        const float xn = fmaxf(sqrtf(snrm[tid][0]), kEPS);
        cosv[b * kS + tid] = sdot[tid][0] / (xn * cn);
    }
}

// ---------------------------------------------------------------------------
// Kernel 2: build bf16 activation matrix G[b, d*5+w].
// ---------------------------------------------------------------------------
__global__ __launch_bounds__(256) void build_a_kernel(const float* __restrict__ x,
                                                      const float* __restrict__ cosv,
                                                      __bf16* __restrict__ G) {
    const int b = blockIdx.y;
    const int d = blockIdx.x * 256 + threadIdx.x;

    __shared__ float cs[kS];
    if (threadIdx.x < kS) cs[threadIdx.x] = cosv[b * kS + threadIdx.x];
    __syncthreads();

    const float* xb = x + (size_t)b * kS * kD + d;
    float t[kS];
#pragma unroll
    for (int s = 0; s < kS; ++s) t[s] = xb[s * kD] * cs[s];

    float g = t[0] + t[1] + t[2] + t[3] + t[4] + t[5];
    __bf16* outp = G + (size_t)b * kK + (size_t)d * kW;
    constexpr float inv6 = 1.0f / 6.0f;
#pragma unroll
    for (int w = 0; w < kW; ++w) {
        outp[w] = (__bf16)(g * inv6);
        if (w < kW - 1) g += t[w + 6] - t[w];
    }
}

// ---------------------------------------------------------------------------
// Kernel 3: out[b,o] = sum_k G[b,k]*conv_w[o,k] + conv_b[o]
// 256 blocks x 4 waves.  Weight stream: 10-stage async global->LDS ring;
// each stage (16 rows x 32 K fp32) is ONE async b128 per wave (waves cover
// disjoint 32B row-slices).  Per superstep (64 K): 1 asynccnt wait + 1 bare
// split barrier, 2 stage refills, 2x(4 WMMA).  8 stages in flight ~ HBM
// latency.  A fragments: register ping-pong from L2-resident G.
// ---------------------------------------------------------------------------
__global__ __launch_bounds__(128) void gemm_kernel(const __bf16* __restrict__ G,
                                                   const float*  __restrict__ Wt,
                                                   const float*  __restrict__ bias,
                                                   float* __restrict__ out) {
    __shared__ float ring[kNS][16][kRowF];

    const int tid  = threadIdx.x;
    const int wave = tid >> 5;
    const int lane = tid & 31;
    const int ln   = lane & 15;
    const int half = lane >> 4;         // 0 | 1
    const int aoff = half * 8;          // A frag: K = aoff + {0..7, 16..23}
    const int boff = half * 8;          // B frag: K = boff + {0..7, 16..23}

    const int n = blockIdx.x * 16 + ln;             // output column (o)
    const int mbase = wave * 64;                    // 4 tiles of 16 batch rows

    // producer mapping: wave w covers bytes [w*32, w*32+32) of each row's
    // 128B stage slice; lane: row = ln, 16B chunk = half
    const unsigned inrow = (unsigned)(wave * 32 + half * 16);
    const char* wgbase = (const char*)(Wt + (size_t)n * kK) + inrow;
    const unsigned lds0 = (unsigned)(uintptr_t)(&ring[0][0][0]);
    const unsigned ldslane = lds0 + (unsigned)ln * (kRowF * 4) + inrow;

    auto issue_b = [&](int buf, int kb) {
        async_load_b128(ldslane + (unsigned)buf * kStageBytes,
                        wgbase + (size_t)kb * 4);
    };

    const __bf16* gbase[4];
#pragma unroll
    for (int mt = 0; mt < 4; ++mt)
        gbase[mt] = G + (size_t)(mbase + mt * 16 + ln) * kK + aoff;

    auto load_a = [&](v8bf (&dst)[4][2], int kb) {
#pragma unroll
        for (int mt = 0; mt < 4; ++mt) {
            const __bf16* ap = gbase[mt] + kb;
            dst[mt][0] = *(const v8bf*)(ap + 0);    // K = aoff + 0..7
            dst[mt][1] = *(const v8bf*)(ap + 16);   // K = aoff + 16..23
        }
    };

    v8f acc[4];
#pragma unroll
    for (int mt = 0; mt < 4; ++mt)
#pragma unroll
        for (int r = 0; r < 8; ++r) acc[mt][r] = 0.0f;

    // consume one ring stage: LDS -> bf16 fragment -> 4 WMMAs
    auto consume = [&](int buf, const v8bf (&acur)[4][2]) {
        const float* bp = &ring[buf][ln][boff];
        const v4f w0 = *(const v4f*)(bp + 0);
        const v4f w1 = *(const v4f*)(bp + 4);
        const v4f w2 = *(const v4f*)(bp + 16);
        const v4f w3 = *(const v4f*)(bp + 20);
        v16bf bfrag;
#pragma unroll
        for (int i = 0; i < 4; ++i) {
            bfrag[i]      = (__bf16)w0[i];
            bfrag[4 + i]  = (__bf16)w1[i];
            bfrag[8 + i]  = (__bf16)w2[i];
            bfrag[12 + i] = (__bf16)w3[i];
        }
#pragma unroll
        for (int mt = 0; mt < 4; ++mt) {
            v16bf afrag;
#pragma unroll
            for (int i = 0; i < 8; ++i) {
                afrag[i]     = acur[mt][0][i];
                afrag[8 + i] = acur[mt][1][i];
            }
            acc[mt] = __builtin_amdgcn_wmma_f32_16x16x32_bf16(
                false, afrag, false, bfrag, (short)0, acc[mt], false, false);
        }
    };

    constexpr int kIters = kK / 32;                 // 640 K-steps
    constexpr int kSupers = kIters / 2;             // 320 supersteps

    // prologue: fill 8 ring stages (asynccnt = 8 per wave) + A for k-step 0
    v8bf aA[4][2], aB[4][2];
#pragma unroll
    for (int st = 0; st < 8; ++st) issue_b(st, st * 32);
    load_a(aA, 0);

    int bufc = 0;   // buffer of first consumed stage (even, 0..8)
    int rf   = 8;   // buffer of first refilled stage (even, 0..8)
    for (int t = 0; t < kSupers; ++t) {
        wait_asynccnt_6();      // my parts of the 2 oldest stages landed
        block_barrier();        // all waves' parts landed (no loadcnt drain)

        // refill 2 stages; their buffers were read in superstep t-1,
        // retired before the barrier above (reuse distance = 8 stages)
        const int k0 = (2 * t + 8) * 32;
        const int k1 = (2 * t + 9) * 32;
        issue_b(rf + 0, (k0 < kK) ? k0 : 0);
        issue_b(rf + 1, (k1 < kK) ? k1 : 0);

        // k-step 2t: prefetch A for 2t+1, compute from aA
        load_a(aB, (2 * t + 1) * 32);
        consume(bufc + 0, aA);

        // k-step 2t+1: prefetch A for 2t+2 (clamped), compute from aB
        const int ka = (2 * t + 2 < kIters) ? (2 * t + 2) * 32 : 0;
        load_a(aA, ka);
        consume(bufc + 1, aB);

        bufc += 2; if (bufc == kNS) bufc = 0;
        rf   += 2; if (rf   == kNS) rf   = 0;
    }

    const float bo = bias[n];
#pragma unroll
    for (int mt = 0; mt < 4; ++mt) {
        const int mrow = mbase + mt * 16 + half * 8;    // D layout: lanes 16-31 -> M+8
#pragma unroll
        for (int r = 0; r < 8; ++r) {
            out[(size_t)(mrow + r) * kO + n] = acc[mt][r] + bo;
        }
    }
}

// ---------------------------------------------------------------------------
extern "C" void kernel_launch(void* const* d_in, const int* in_sizes, int n_in,
                              void* d_out, int out_size, void* d_ws, size_t ws_size,
                              hipStream_t stream) {
    const float* x      = (const float*)d_in[0];
    const float* conv_w = (const float*)d_in[1];
    const float* conv_b = (const float*)d_in[2];
    float* out = (float*)d_out;

    // workspace: [0,10KB) cos values ; [16KB, 16KB+10.5MB) bf16 activation G
    float*  cosv = (float*)d_ws;
    __bf16* G    = (__bf16*)((char*)d_ws + 16384);

    stats_kernel<<<kB, 256, 0, stream>>>(x, cosv);
    build_a_kernel<<<dim3(kD / 256, kB), 256, 0, stream>>>(x, cosv, G);
    gemm_kernel<<<kO / 16, 128, 0, stream>>>(G, conv_w, conv_b, out);
}
